// VectorQuantizerV2_90288802496519
// MI455X (gfx1250) — matrix operand order repaired and checked by reading, not compiled, not run on hardware
//
#include <hip/hip_runtime.h>
#include <hip/hip_bf16.h>

// ---------------- problem constants ----------------
#define T_ROWS   16384          // B*N = 8*2048
#define DIM      256
#define KCODES   8192
#define NTILES   (KCODES / 16)
#define EMA_DEC  0.99f
#define EMA_ONE  0.01f
#define EPSV     1e-5f

typedef __attribute__((ext_vector_type(16))) _Float16 v16h;
typedef __attribute__((ext_vector_type(8)))  float    v8f;

#if __has_builtin(__builtin_amdgcn_global_load_async_to_lds_b128)
#define USE_ASYNC 1
#else
#define USE_ASYNC 0
#endif

#if USE_ASYNC
// clang expects: (int4 AS1* gsrc, int4 AS3* ldst, imm offset, imm cpol)
typedef int v4i_t __attribute__((vector_size(16)));
typedef __attribute__((address_space(1))) v4i_t* gptr_t;
typedef __attribute__((address_space(3))) v4i_t* lptr_t;
__device__ __forceinline__ void wait_async0() {
#if __has_builtin(__builtin_amdgcn_s_wait_asynccnt)
    __builtin_amdgcn_s_wait_asynccnt(0);
#else
    asm volatile("s_wait_asynccnt 0x0" ::: "memory");
#endif
}
#endif

// ---------------- workspace layout (bytes) ----------------
//  [0, 4194304)              emb_h   : K*D f16 codebook
//  [4194304, +32768)         enorm   : ||e_k||^2  (f32, K)
//  [4227072, +32768)         cluster : per-code counts (f32, K, atomic)
//  [4259840, +16)            scal    : [0]=sum_sq [1]=n_sum [2]=ent [3]=unique
#define WS_ENORM   4194304
#define WS_CLUST   4227072
#define WS_SCAL    4259840

// ---------------- output layout (floats) ----------------
#define O_ZQ     0           // 4194304
#define O_IDX    4194304     // 16384
#define O_SCAL   4210688     // vq_loss, perplexity, usage_ratio, unique
#define O_EMB    4210692     // 2097152
#define O_ECS    6307844     // 8192
#define O_EMAW   6316036     // 2097152
#define O_USAGE  8413188     // 8192

// =====================================================================
// Kernel 0: per-call init.
// =====================================================================
__global__ __launch_bounds__(256) void vq_init(const float* __restrict__ emaw_in,
                                               float* __restrict__ emaw_out,
                                               float* __restrict__ cluster,
                                               float* __restrict__ scal)
{
    const size_t i = (size_t)blockIdx.x * 256 + threadIdx.x;   // covers K*D
    emaw_out[i] = EMA_DEC * emaw_in[i];
    if (i < KCODES) cluster[i] = 0.0f;
    if (i < 4)      scal[i]    = 0.0f;
}

// =====================================================================
// Kernel 1: codebook prep: fp32 -> f16 + ||e||^2 per code row.
// =====================================================================
__global__ __launch_bounds__(256) void vq_prep(const float* __restrict__ emb,
                                               _Float16* __restrict__ emb_h,
                                               float* __restrict__ enorm)
{
    const int wave = threadIdx.x >> 5;
    const int lane = threadIdx.x & 31;
    const int k    = blockIdx.x * 8 + wave;
    const float* er = emb + (size_t)k * DIM;
    const int c0 = lane * 8;

    __align__(16) _Float16 h[8];
    float ss = 0.0f;
#pragma unroll
    for (int j = 0; j < 8; ++j) {
        const float v = er[c0 + j];
        ss += v * v;
        h[j] = (_Float16)v;
    }
    *(uint4*)(emb_h + (size_t)k * DIM + c0) = *(const uint4*)h;
#pragma unroll
    for (int m = 16; m >= 1; m >>= 1) ss += __shfl_xor(ss, m, 32);
    if (lane == 0) enorm[k] = ss;
}

// =====================================================================
// Kernel 2: fused distance GEMM (f16 WMMA, f32 accum) + argmin + epilogue.
// Block = 128 thr = 4 waves; wave owns 32 rows (two A-tile sets).
// Async path: double-buffered LDS fed by GLOBAL_LOAD_ASYNC_TO_LDS_B128
// (no data VGPRs -> no spills), one barrier per tile.
// =====================================================================
__global__ __launch_bounds__(128) void vq_main(const float*     __restrict__ z,
                                               const float*     __restrict__ emb,
                                               const _Float16*  __restrict__ emb_h,
                                               const float*     __restrict__ enorm,
                                               float* __restrict__ cluster,
                                               float* __restrict__ scal,
                                               float* __restrict__ emaw_out,
                                               float* __restrict__ zq_out,
                                               float* __restrict__ idx_out)
{
#if USE_ASYNC
    __shared__ __align__(32) _Float16 sB[2][16 * DIM];   // 2 x 8 KB code tile
    __shared__ float sE[2][16];
#else
    __shared__ __align__(32) _Float16 sB[1][16 * DIM];   // 8 KB code tile
    __shared__ float sE[1][16];
#endif

    const int tid  = threadIdx.x;
    const int wave = tid >> 5;
    const int lane = tid & 31;
    const int n    = lane & 15;        // column within 16-wide tile
    const int hi   = lane >> 4;        // lane half
    const int rowbase = blockIdx.x * 128 + wave * 32;

    // ---- A fragments: 2 x (16 rows x 256), fp32 -> f16, VGPR-resident ----
    const float* zr0 = z + (size_t)(rowbase + n) * DIM;
    const float* zr1 = z + (size_t)(rowbase + 16 + n) * DIM;
    v16h a0[8], a1[8];
#pragma unroll
    for (int kc = 0; kc < 8; ++kc) {
        const int o1 = kc * 32 + hi * 16;
        const int o2 = o1 + 8;
#pragma unroll
        for (int j = 0; j < 8; ++j) {
            a0[kc][j]     = (_Float16)zr0[o1 + j];
            a0[kc][j + 8] = (_Float16)zr0[o2 + j];
            a1[kc][j]     = (_Float16)zr1[o1 + j];
            a1[kc][j + 8] = (_Float16)zr1[o2 + j];
        }
    }

    float minv0[8], minv1[8];
    int   mini0[8], mini1[8];
#pragma unroll
    for (int v = 0; v < 8; ++v) {
        minv0[v] = 3.4e38f; mini0[v] = 0;
        minv1[v] = 3.4e38f; mini1[v] = 0;
    }

#if USE_ASYNC
    // ---- prologue: async-stage tile 0 (global -> LDS, no data VGPRs) ----
    {
        gptr_t g = (gptr_t)emb_h;
        lptr_t l = (lptr_t)&sB[0][0];
#pragma unroll
        for (int i = 0; i < 4; ++i)
            __builtin_amdgcn_global_load_async_to_lds_b128(
                g + (tid + 128 * i), l + (tid + 128 * i), 0, 0);
        if (tid < 16) sE[0][tid] = enorm[tid];
        wait_async0();
    }
    __syncthreads();
#endif

    // ---- sweep all 8192 codes in tiles of 16 ----
    for (int t = 0; t < NTILES; ++t) {
#if USE_ASYNC
        const int buf = t & 1;
        // kick off async copy of tile t+1 into the other buffer
        float ne = 0.0f;
        if (t + 1 < NTILES) {
            gptr_t g = (gptr_t)(emb_h + (size_t)(t + 1) * 16 * DIM);
            lptr_t l = (lptr_t)&sB[buf ^ 1][0];
#pragma unroll
            for (int i = 0; i < 4; ++i)
                __builtin_amdgcn_global_load_async_to_lds_b128(
                    g + (tid + 128 * i), l + (tid + 128 * i), 0, 0);
            if (tid < 16) ne = enorm[(t + 1) * 16 + tid];
        }
#else
        const int buf = 0;
        // synchronous staging (short-lived regs, two barriers)
        {
            const uint4* src = (const uint4*)(emb_h + (size_t)t * 16 * DIM);
            uint4* dst = (uint4*)&sB[0][0];
#pragma unroll
            for (int i = 0; i < 4; ++i) dst[tid + 128 * i] = src[tid + 128 * i];
            if (tid < 16) sE[0][tid] = enorm[t * 16 + tid];
        }
        __syncthreads();
#endif

        const _Float16* sbuf = &sB[buf][0];
        v8f acc0 = {}, acc1 = {};
#pragma unroll
        for (int kc = 0; kc < 8; ++kc) {
            const v16h b = *(const v16h*)(sbuf + n * DIM + kc * 32 + hi * 16);
            acc0 = __builtin_amdgcn_wmma_f32_16x16x32_f16(
                       false, a0[kc], false, b, (short)0, acc0, false, false);
            acc1 = __builtin_amdgcn_wmma_f32_16x16x32_f16(
                       false, a1[kc], false, b, (short)0, acc1, false, false);
        }

        const float en   = sE[buf][n];
        const int   code = t * 16 + n;
#pragma unroll
        for (int v = 0; v < 8; ++v) {
            const float d0 = en - 2.0f * acc0[v];  // ||z||^2 dropped (row-const)
            const float d1 = en - 2.0f * acc1[v];
            if (d0 < minv0[v]) { minv0[v] = d0; mini0[v] = code; }
            if (d1 < minv1[v]) { minv1[v] = d1; mini1[v] = code; }
        }

#if USE_ASYNC
        if (t + 1 < NTILES && tid < 16) sE[buf ^ 1][tid] = ne;
        wait_async0();          // tile t+1 fully landed in LDS
        __syncthreads();        // one barrier per tile
#else
        __syncthreads();
#endif
    }

    // ---- argmin across the 16 column-lanes (stays within each half) ----
#pragma unroll
    for (int v = 0; v < 8; ++v) {
#pragma unroll
        for (int m = 1; m <= 8; m <<= 1) {
            float ov = __shfl_xor(minv0[v], m, 32);
            int   oi = __shfl_xor(mini0[v], m, 32);
            if (ov < minv0[v] || (ov == minv0[v] && oi < mini0[v])) {
                minv0[v] = ov; mini0[v] = oi;
            }
            ov = __shfl_xor(minv1[v], m, 32);
            oi = __shfl_xor(mini1[v], m, 32);
            if (ov < minv1[v] || (ov == minv1[v] && oi < mini1[v])) {
                minv1[v] = ov; mini1[v] = oi;
            }
        }
    }
    int rowidx[32];
#pragma unroll
    for (int v = 0; v < 8; ++v) {
        rowidx[v]      = __shfl(mini0[v], 0, 32);
        rowidx[v + 8]  = __shfl(mini0[v], 16, 32);
        rowidx[v + 16] = __shfl(mini1[v], 0, 32);
        rowidx[v + 24] = __shfl(mini1[v], 16, 32);
    }

    // ---- epilogue: z_q gather, straight-through out, EMA scatter, loss ----
    float local_ss = 0.0f;
#pragma unroll 1
    for (int r = 0; r < 32; ++r) {
        const int code = rowidx[r];
        const int grow = rowbase + r;
        const float* zrow = z   + (size_t)grow * DIM;
        const float* erow = emb + (size_t)code * DIM;
        float*       qrow = zq_out   + (size_t)grow * DIM;
        float*       wrow = emaw_out + (size_t)code * DIM;
        const int col0 = lane * 8;
#pragma unroll
        for (int j = 0; j < 8; ++j) {
            const float zv = zrow[col0 + j];
            const float ev = erow[col0 + j];
            qrow[col0 + j] = zv + (ev - zv);          // straight-through value
            const float d = zv - ev;
            local_ss += d * d;
            atomicAdd(&wrow[col0 + j], EMA_ONE * zv); // 0.01 * dw scatter
        }
        if (lane == 0) {
            idx_out[grow] = (float)code;
            atomicAdd(&cluster[code], 1.0f);
        }
    }
#pragma unroll
    for (int m = 16; m >= 1; m >>= 1) local_ss += __shfl_xor(local_ss, m, 32);
    if (lane == 0) atomicAdd(&scal[0], local_ss);
}

// =====================================================================
// Kernel 3: global reductions over K (n_sum, entropy, unique count).
// =====================================================================
__global__ __launch_bounds__(256) void vq_reduce(const float* __restrict__ ecs_in,
                                                 const float* __restrict__ cluster,
                                                 const float* __restrict__ usage_in,
                                                 float* __restrict__ scal)
{
    const int k    = blockIdx.x * 256 + threadIdx.x;
    const int lane = threadIdx.x & 31;
    const float cs  = cluster[k];
    float raw = EMA_DEC * ecs_in[k] + EMA_ONE * cs;
    float ind = (usage_in[k] + cs > 0.0f) ? 1.0f : 0.0f;
    const float p = cs * (1.0f / (float)T_ROWS);
    float ent = -p * __logf(p + 1e-10f);
#pragma unroll
    for (int m = 16; m >= 1; m >>= 1) {
        raw += __shfl_xor(raw, m, 32);
        ind += __shfl_xor(ind, m, 32);
        ent += __shfl_xor(ent, m, 32);
    }
    if (lane == 0) {
        atomicAdd(&scal[1], raw);
        atomicAdd(&scal[3], ind);
        atomicAdd(&scal[2], ent);
    }
}

// =====================================================================
// Kernel 4: finalize — Laplace smoothing, codebook divide, scalars.
// =====================================================================
__global__ __launch_bounds__(256) void vq_fin(const float* __restrict__ ecs_in,
                                              const float* __restrict__ cluster,
                                              const float* __restrict__ usage_in,
                                              const float* __restrict__ scal,
                                              const float* __restrict__ emaw_out,
                                              float* __restrict__ emb_out,
                                              float* __restrict__ ecs_out,
                                              float* __restrict__ usage_out,
                                              float* __restrict__ scalars_out)
{
    const int wave = threadIdx.x >> 5;
    const int lane = threadIdx.x & 31;
    const int k    = blockIdx.x * 8 + wave;

    const float cs   = cluster[k];
    const float raw  = EMA_DEC * ecs_in[k] + EMA_ONE * cs;
    const float nsum = scal[1];
    const float ecs  = (raw + EPSV) / (nsum + (float)KCODES * EPSV) * nsum;
    const float den  = fmaxf(ecs, EPSV);
    const float rden = 1.0f / den;

    const int c0 = lane * 8;
    const size_t base = (size_t)k * DIM + c0;
#pragma unroll
    for (int j = 0; j < 8; ++j)
        emb_out[base + j] = emaw_out[base + j] * rden;

    if (lane == 0) {
        ecs_out[k]   = ecs;
        usage_out[k] = usage_in[k] + cs;
    }
    if (blockIdx.x == 0 && threadIdx.x == 0) {
        scalars_out[0] = 1.25f * scal[0] * (1.0f / (float)(T_ROWS * DIM)); // vq_loss
        scalars_out[1] = __expf(scal[2]);                                  // perplexity
        scalars_out[2] = scal[3] * (1.0f / (float)KCODES);                 // usage_ratio
        scalars_out[3] = scal[3];                                          // unique_codes
    }
}

// =====================================================================
extern "C" void kernel_launch(void* const* d_in, const int* in_sizes, int n_in,
                              void* d_out, int out_size, void* d_ws, size_t ws_size,
                              hipStream_t stream)
{
    const float* z        = (const float*)d_in[0];
    const float* emb      = (const float*)d_in[1];
    const float* ecs_in   = (const float*)d_in[2];
    const float* emaw_in  = (const float*)d_in[3];
    const float* usage_in = (const float*)d_in[4];

    float* out         = (float*)d_out;
    float* zq_out      = out + O_ZQ;
    float* idx_out     = out + O_IDX;
    float* scalars_out = out + O_SCAL;
    float* emb_out     = out + O_EMB;
    float* ecs_out     = out + O_ECS;
    float* emaw_out    = out + O_EMAW;
    float* usage_out   = out + O_USAGE;

    char* ws = (char*)d_ws;
    _Float16* emb_h  = (_Float16*)ws;
    float*    enorm  = (float*)(ws + WS_ENORM);
    float*    clus   = (float*)(ws + WS_CLUST);
    float*    scal   = (float*)(ws + WS_SCAL);

    vq_init  <<<(KCODES * DIM) / 256, 256, 0, stream>>>(emaw_in, emaw_out, clus, scal);
    vq_prep  <<<KCODES / 8,           256, 0, stream>>>(emb, emb_h, enorm);
    vq_main  <<<T_ROWS / 128,         128, 0, stream>>>(z, emb, emb_h, enorm,
                                                        clus, scal, emaw_out,
                                                        zq_out, idx_out);
    vq_reduce<<<KCODES / 256,         256, 0, stream>>>(ecs_in, clus, usage_in, scal);
    vq_fin   <<<KCODES / 8,           256, 0, stream>>>(ecs_in, clus, usage_in, scal,
                                                        emaw_out, emb_out, ecs_out,
                                                        usage_out, scalars_out);
}
